// DiffMoeMLP_55379308315216
// MI455X (gfx1250) — matrix operand order repaired and checked by reading, not compiled, not run on hardware
//
#include <hip/hip_runtime.h>
#include <hip/hip_bf16.h>
#include <math.h>

#define D_EMB 1024
#define D_FF  4096
#define NEXP  8
#define NTOK  8192
#define KCAP  1024

typedef __attribute__((ext_vector_type(16))) _Float16 v16h;
typedef __attribute__((ext_vector_type(8)))  _Float16 v8h;
typedef __attribute__((ext_vector_type(4)))  _Float16 h4;
typedef __attribute__((ext_vector_type(8)))  float    v8f;
typedef __attribute__((ext_vector_type(4)))  float    f32x4;

static __device__ __forceinline__ float gelu_tanh(float x) {
  float t = 0.7978845608028654f * (x + 0.044715f * x * x * x);
  return 0.5f * x * (1.0f + tanhf(t));
}

// A fragment (M=16, K=32) f16 row-major (K contiguous), leading dim lda.
// A layout (matches B): lanes 0-15 = row r, K 0..15; lanes 16-31 = K 16..31.
static __device__ __forceinline__ v16h load_a_f16(const _Float16* A, int lda, int lane) {
  int r = lane & 15;
  int g = (lane >> 4) * 16;
  const _Float16* p = A + (size_t)r * lda + g;
  v8h lo = *(const v8h*)p;          // K = g .. g+7
  v8h hi = *(const v8h*)(p + 8);    // K = g+8 .. g+15
  return __builtin_shufflevector(lo, hi, 0, 1, 2, 3, 4, 5, 6, 7,
                                 8, 9, 10, 11, 12, 13, 14, 15);
}

// B fragment (K=32, N=16) f16 stored N-major (row = output col, K contiguous).
// B layout used here (matches A): lanes 0-15 = col n, K 0..15; lanes 16-31 = col n, K 16..31 (contiguous 32B).
static __device__ __forceinline__ v16h load_b_f16(const _Float16* B, int ldb, int lane) {
  int n = lane & 15;
  int g = (lane >> 4) * 16;
  return *(const v16h*)(B + (size_t)n * ldb + g);
}

#define WMMA_F16(A, B, C) \
  __builtin_amdgcn_wmma_f32_16x16x32_f16(false, (A), false, (B), (short)0, (C), false, false)

// ---------------- K0: out = x ----------------
__global__ __launch_bounds__(256) void k_copy(const float* __restrict__ x,
                                              float* __restrict__ out) {
  size_t i = (size_t)blockIdx.x * 256 + threadIdx.x;
  ((f32x4*)out)[i] = ((const f32x4*)x)[i];
}

// ---------------- K0b: fp32 weights -> f16 (v_cvt_pk_rtz_f16_f32) ----------------
__global__ __launch_bounds__(256) void k_cvt(const float* __restrict__ src,
                                             _Float16* __restrict__ dst) {
  size_t i = (size_t)blockIdx.x * 256 + threadIdx.x;
  f32x4 q = ((const f32x4*)src)[i];
  auto p0 = __builtin_amdgcn_cvt_pkrtz(q[0], q[1]);
  auto p1 = __builtin_amdgcn_cvt_pkrtz(q[2], q[3]);
  h4 o;
  o[0] = p0[0]; o[1] = p0[1]; o[2] = p1[0]; o[3] = p1[1];
  ((h4*)dst)[i] = o;
}

// ---------------- K1: gating scores (8192 x 8) ----------------
__global__ __launch_bounds__(256) void k_gate(const float* __restrict__ x,
                                              const float* __restrict__ gw,
                                              float* __restrict__ scores) {
  __shared__ float sgw[NEXP * D_EMB];  // 32 KB
  for (int i = threadIdx.x; i < NEXP * D_EMB; i += 256) sgw[i] = gw[i];
  __syncthreads();
  int lane = threadIdx.x & 31;
  int token = blockIdx.x * 8 + (threadIdx.x >> 5);
  const float* xr = x + (size_t)token * D_EMB;
  float acc[NEXP] = {};
  for (int i = lane; i < D_EMB; i += 32) {
    float xv = xr[i];
#pragma unroll
    for (int e = 0; e < NEXP; ++e) acc[e] += xv * sgw[e * D_EMB + i];
  }
#pragma unroll
  for (int e = 0; e < NEXP; ++e) {
    float v = acc[e];
    for (int off = 16; off > 0; off >>= 1) v += __shfl_xor(v, off, 32);
    if (lane == 0) scores[(size_t)token * NEXP + e] = 0.5f * (tanhf(v) + 1.0f);
  }
}

// ---------------- K2: per-expert top-1024 via LDS bitonic sort ----------------
__global__ __launch_bounds__(1024) void k_topk(const float* __restrict__ scores,
                                               int* __restrict__ kept_idx,
                                               float* __restrict__ kept_w) {
  __shared__ unsigned long long keys[NTOK];  // 64 KB of the 320 KB WGP LDS
  int e = blockIdx.x;
  int tid = threadIdx.x;
  for (int t = tid; t < NTOK; t += 1024) {
    unsigned sb = __float_as_uint(scores[(size_t)t * NEXP + e]);  // scores in [0,1]: bits monotonic
    keys[t] = ((unsigned long long)sb << 32) | (unsigned)(NTOK - 1 - t);  // ties: smaller token wins
  }
  __syncthreads();
  for (unsigned size = 2; size <= NTOK; size <<= 1) {
    for (unsigned stride = size >> 1; stride > 0; stride >>= 1) {
#pragma unroll 1
      for (int v = 0; v < NTOK / 2 / 1024; ++v) {
        unsigned t = (unsigned)tid + v * 1024u;
        unsigned lo = ((t & ~(stride - 1)) << 1) | (t & (stride - 1));
        unsigned hi = lo + stride;
        bool desc = ((lo & size) == 0);
        unsigned long long a = keys[lo], b = keys[hi];
        if (desc ? (a < b) : (a > b)) { keys[lo] = b; keys[hi] = a; }
      }
      __syncthreads();
    }
  }
  unsigned long long kk = keys[tid];  // first KCAP entries = descending top-k
  kept_idx[e * KCAP + tid] = (NTOK - 1) - (int)(kk & 0xFFFFFFFFu);
  kept_w[e * KCAP + tid] = __uint_as_float((unsigned)(kk >> 32));
}

// ---------------- K3: gather + LayerNorm -> f16 A operand ----------------
__global__ __launch_bounds__(256) void k_gather_ln(const float* __restrict__ x,
                                                   const int* __restrict__ kept_idx,
                                                   const float* __restrict__ gamma,
                                                   const float* __restrict__ beta,
                                                   _Float16* __restrict__ yln) {
  int lane = threadIdx.x & 31;
  int row = blockIdx.x * 8 + (threadIdx.x >> 5);  // row = e*KCAP + j
  int token = kept_idx[row];
  const float* xr = x + (size_t)token * D_EMB;
  float vals[32];
  float s = 0.f, s2 = 0.f;
#pragma unroll
  for (int i = 0; i < 32; ++i) {
    float v = xr[lane + i * 32];
    vals[i] = v; s += v; s2 += v * v;
  }
  for (int off = 16; off > 0; off >>= 1) {
    s  += __shfl_xor(s, off, 32);
    s2 += __shfl_xor(s2, off, 32);
  }
  float mu = s * (1.0f / D_EMB);
  float var = s2 * (1.0f / D_EMB) - mu * mu;
  float rstd = rsqrtf(var + 1e-5f);
  _Float16* orow = yln + (size_t)row * D_EMB;
#pragma unroll
  for (int i = 0; i < 32; ++i) {
    int c = lane + i * 32;
    orow[c] = (_Float16)((vals[i] - mu) * rstd * gamma[c] + beta[c]);
  }
}

// ---------------- K4: fc1 (per-expert 1024x1024 @ 1024x4096^T) + bias + GELU ----------------
// 32x64 output per wave: 2 A-frags x 4 B-frags -> 8 WMMAs per k-step.
__global__ __launch_bounds__(256)
void k_fc1(const _Float16* __restrict__ yln, const _Float16* __restrict__ w1,
           const float* __restrict__ b1s, _Float16* __restrict__ h) {
  int lane = threadIdx.x & 31;
  int unit = blockIdx.x * 8 + (threadIdx.x >> 5);
  int e  = unit >> 11;          // 32*64 units per expert
  int mt = (unit >> 6) & 31;    // 32-row M tile
  int ns = unit & 63;           // 64-col N strip
  const _Float16* A = yln + ((size_t)e * KCAP + (size_t)mt * 32) * D_EMB;
  const _Float16* B = w1 + ((size_t)e * D_FF + (size_t)ns * 64) * D_EMB;
  v8f c00 = {}, c01 = {}, c02 = {}, c03 = {};
  v8f c10 = {}, c11 = {}, c12 = {}, c13 = {};
  for (int k = 0; k < D_EMB; k += 32) {
    v16h a0 = load_a_f16(A + k, D_EMB, lane);
    v16h a1 = load_a_f16(A + 16 * D_EMB + k, D_EMB, lane);
    v16h b0 = load_b_f16(B + k,              D_EMB, lane);
    v16h b1 = load_b_f16(B + 16 * D_EMB + k, D_EMB, lane);
    v16h b2 = load_b_f16(B + 32 * D_EMB + k, D_EMB, lane);
    v16h b3 = load_b_f16(B + 48 * D_EMB + k, D_EMB, lane);
    c00 = WMMA_F16(a0, b0, c00);
    c01 = WMMA_F16(a0, b1, c01);
    c02 = WMMA_F16(a0, b2, c02);
    c03 = WMMA_F16(a0, b3, c03);
    c10 = WMMA_F16(a1, b0, c10);
    c11 = WMMA_F16(a1, b1, c11);
    c12 = WMMA_F16(a1, b2, c12);
    c13 = WMMA_F16(a1, b3, c13);
    __builtin_prefetch((const void*)(B + k + 32), 0, 1);
  }
  int n0 = lane & 15;
  int mo = (lane >> 4) * 8;
  const float* bias = b1s + (size_t)e * D_FF + ns * 64 + n0;
  _Float16* hbase = h + (size_t)e * KCAP * D_FF + ns * 64 + n0;
#pragma unroll
  for (int mtile = 0; mtile < 2; ++mtile) {
    _Float16* hp = hbase + (size_t)(mt * 32 + mtile * 16 + mo) * D_FF;
#pragma unroll
    for (int t = 0; t < 4; ++t) {
      v8f acc = (mtile == 0)
                  ? ((t == 0) ? c00 : (t == 1) ? c01 : (t == 2) ? c02 : c03)
                  : ((t == 0) ? c10 : (t == 1) ? c11 : (t == 2) ? c12 : c13);
      float bb = bias[t * 16];
#pragma unroll
      for (int i = 0; i < 8; ++i)
        hp[(size_t)i * D_FF + t * 16] = (_Float16)gelu_tanh(acc[i] + bb);
    }
  }
}

// ---------------- K5: fc2 (per-expert 1024x4096 @ 4096x1024^T) + bias, *gate, scatter-add ----------------
__global__ __launch_bounds__(256)
void k_fc2(const _Float16* __restrict__ h, const _Float16* __restrict__ w2,
           const float* __restrict__ b2s, const int* __restrict__ kept_idx,
           const float* __restrict__ kept_w, float* __restrict__ out) {
  int lane = threadIdx.x & 31;
  int unit = blockIdx.x * 8 + (threadIdx.x >> 5);
  int e  = unit >> 9;           // 32*16 units per expert
  int mt = (unit >> 4) & 31;    // 32-row M tile
  int ns = unit & 15;           // 64-col N strip
  const _Float16* A = h + ((size_t)e * KCAP + (size_t)mt * 32) * D_FF;
  const _Float16* B = w2 + ((size_t)e * D_EMB + (size_t)ns * 64) * D_FF;
  v8f c00 = {}, c01 = {}, c02 = {}, c03 = {};
  v8f c10 = {}, c11 = {}, c12 = {}, c13 = {};
  for (int k = 0; k < D_FF; k += 32) {
    v16h a0 = load_a_f16(A + k, D_FF, lane);
    v16h a1 = load_a_f16(A + 16 * D_FF + k, D_FF, lane);
    v16h b0 = load_b_f16(B + k,             D_FF, lane);
    v16h b1 = load_b_f16(B + 16 * D_FF + k, D_FF, lane);
    v16h b2 = load_b_f16(B + 32 * D_FF + k, D_FF, lane);
    v16h b3 = load_b_f16(B + 48 * D_FF + k, D_FF, lane);
    c00 = WMMA_F16(a0, b0, c00);
    c01 = WMMA_F16(a0, b1, c01);
    c02 = WMMA_F16(a0, b2, c02);
    c03 = WMMA_F16(a0, b3, c03);
    c10 = WMMA_F16(a1, b0, c10);
    c11 = WMMA_F16(a1, b1, c11);
    c12 = WMMA_F16(a1, b2, c12);
    c13 = WMMA_F16(a1, b3, c13);
    __builtin_prefetch((const void*)(B + k + 32), 0, 1);
  }
  int n0 = lane & 15;
  int mo = (lane >> 4) * 8;
#pragma unroll
  for (int mtile = 0; mtile < 2; ++mtile) {
    int jbase = mt * 32 + mtile * 16 + mo;
#pragma unroll
    for (int i = 0; i < 8; ++i) {
      int j = jbase + i;
      int token = kept_idx[e * KCAP + j];
      float w = kept_w[e * KCAP + j];
      float* orow = out + (size_t)token * D_EMB + ns * 64 + n0;
#pragma unroll
      for (int t = 0; t < 4; ++t) {
        v8f acc = (mtile == 0)
                    ? ((t == 0) ? c00 : (t == 1) ? c01 : (t == 2) ? c02 : c03)
                    : ((t == 0) ? c10 : (t == 1) ? c11 : (t == 2) ? c12 : c13);
        float bias = b2s[(size_t)e * D_EMB + ns * 64 + t * 16 + n0];
        atomicAdd(orow + t * 16, (acc[i] + bias) * w);  // duplicate tokens -> accumulate
      }
    }
  }
}

extern "C" void kernel_launch(void* const* d_in, const int* in_sizes, int n_in,
                              void* d_out, int out_size, void* d_ws, size_t ws_size,
                              hipStream_t stream) {
  (void)in_sizes; (void)n_in; (void)out_size; (void)ws_size;
  const float* x      = (const float*)d_in[0];
  const float* gate_w = (const float*)d_in[1];
  const float* ln_g   = (const float*)d_in[2];
  const float* ln_b   = (const float*)d_in[3];
  const float* fc1s   = (const float*)d_in[4];
  const float* b1s    = (const float*)d_in[5];
  const float* fc2s   = (const float*)d_in[6];
  const float* b2s    = (const float*)d_in[7];
  float* out = (float*)d_out;

  char* ws = (char*)d_ws;
  const size_t MB = 1u << 20;
  float*    scores   = (float*)(ws);                           // 256 KB
  int*      kept_idx = (int*)(ws + (size_t)(256 << 10));       // 32 KB
  float*    kept_w   = (float*)(ws + (size_t)(288 << 10));     // 32 KB
  _Float16* yln      = (_Float16*)(ws + 1 * MB);               // 16 MB
  _Float16* h        = (_Float16*)(ws + 17 * MB);              // 64 MB
  _Float16* w1h      = (_Float16*)(ws + 81 * MB);              // 64 MB
  _Float16* w2h      = (_Float16*)(ws + 145 * MB);             // 64 MB (total 209 MB)

  const int NW = NEXP * D_FF * D_EMB;  // weight element count per matrix set

  k_copy<<<(NTOK * D_EMB / 4) / 256, 256, 0, stream>>>(x, out);
  k_cvt<<<(NW / 4) / 256, 256, 0, stream>>>(fc1s, w1h);
  k_cvt<<<(NW / 4) / 256, 256, 0, stream>>>(fc2s, w2h);
  k_gate<<<NTOK / 8, 256, 0, stream>>>(x, gate_w, scores);
  k_topk<<<NEXP, 1024, 0, stream>>>(scores, kept_idx, kept_w);
  k_gather_ln<<<NTOK / 8, 256, 0, stream>>>(x, kept_idx, ln_g, ln_b, yln);
  k_fc1<<<(NEXP * 32 * 64) / 8, 256, 0, stream>>>(yln, w1h, b1s, h);
  k_fc2<<<(NEXP * 32 * 16) / 8, 256, 0, stream>>>(h, w2h, b2s, kept_idx, kept_w, out);
}
